// WarehouseDiffusionModel_59270548685084
// MI455X (gfx1250) — matrix operand order repaired and checked by reading, not compiled, not run on hardware
//
#include <hip/hip_runtime.h>
#include <hip/hip_bf16.h>
#include <math.h>

// ---------------------------------------------------------------------------
// Warehouse diffusion GNN forward for MI455X (gfx1250, wave32, WMMA).
//
// Key restructuring: the 97-wide message-MLP input concat(h[dst],h[src],
// radial,temb) is algebraically split, so the [E,97]@[97,32] GEMM becomes
// per-node [8192,32]@[32,32] precomputes (Hd, Hs) + per-edge adds. The
// remaining per-edge 32->32 GEMMs (msg2, pos1) run on
// v_wmma_f32_16x16x32_f16, computed transposed (Y^T = W^T @ M^T) so that
// lane == edge column: attention/pos dots are lane-local + one shfl_xor(16),
// and the D->B relayout between the chained WMMAs needs only partner-lane
// (lane^16) exchanges.
//
// This revision removes the IEEE-division expansions: silu/sigmoid use the
// hardware v_rcp_f32 (__builtin_amdgcn_rcpf) and v_sqrt_f32, cutting the
// per-tile VALU stream ~3x so the WMMA (XDL/TRANS) pipe is the limiter.
// ---------------------------------------------------------------------------

typedef __attribute__((ext_vector_type(16))) _Float16 v16h;
typedef __attribute__((ext_vector_type(8)))  float    v8f;

#define BGR 64          // graphs
#define NG  16          // goal nodes
#define NSH 112         // shelf nodes
#define NN  128         // nodes per graph
#define NT  (BGR * NN)  // 8192 total nodes
#define DD  32          // embedding dim
#define NL  5           // layers

static __device__ __forceinline__ float frcp(float x) {
  return __builtin_amdgcn_rcpf(x);               // single v_rcp_f32
}
static __device__ __forceinline__ float fsqrt(float x) {
  return __builtin_amdgcn_sqrtf(x);              // single v_sqrt_f32
}
static __device__ __forceinline__ float sigmf(float x) {
  return frcp(1.0f + __expf(-x));
}
static __device__ __forceinline__ float siluf(float x) {
  return x * frcp(1.0f + __expf(-x));
}

static __device__ __forceinline__ v8f wmma16(v16h a, v16h b, v8f c) {
  // D = A(16x32 f16) x B(32x16 f16) + C(16x16 f32)
  return __builtin_amdgcn_wmma_f32_16x16x32_f16(false, a, false, b,
                                                (short)0, c, false, false);
}

// --------------------------- init kernels ----------------------------------

__global__ void k_init_temb(const int* __restrict__ ts, float* __restrict__ temb) {
  int tid = blockIdx.x * blockDim.x + threadIdx.x;
  if (tid >= BGR * DD) return;
  int b = tid >> 5, f = tid & 31;
  float t = (float)ts[b];
  int k = (f < 16) ? f : (f - 16);
  float freq = __expf(-logf(10000.0f) * (float)k / 16.0f);
  float ang = t * freq;
  temb[tid] = (f < 16) ? cosf(ang) : sinf(ang);
}

__global__ void k_init_hp(const float* __restrict__ W_hin, const float* __restrict__ b_hin,
                          const float* __restrict__ goal_pos, const float* __restrict__ pos,
                          float* __restrict__ h, float* __restrict__ p) {
  int tid = blockIdx.x * blockDim.x + threadIdx.x;
  if (tid >= NT) return;
  int b = tid >> 7, ii = tid & 127;
  int row = (ii < NG) ? 0 : 1;            // one-hot [1,0] goals / [0,1] shelves
  #pragma unroll
  for (int f = 0; f < DD; ++f) h[tid * DD + f] = W_hin[row * DD + f] + b_hin[f];
  float px, py;
  if (ii < NG) { px = goal_pos[ii * 2]; py = goal_pos[ii * 2 + 1]; }
  else { int s = ii - NG; px = pos[(b * NSH + s) * 2]; py = pos[(b * NSH + s) * 2 + 1]; }
  p[tid * 2] = px; p[tid * 2 + 1] = py;
}

// ------------------ per-layer node-level precomputes -----------------------

// Hd = H @ W_msg1[rows 0:32], Hs = H @ W_msg1[rows 32:64]
__global__ void k_precomp(const float* __restrict__ h, const float* __restrict__ Wm1,
                          float* __restrict__ Hd, float* __restrict__ Hs) {
  int tid = blockIdx.x * blockDim.x + threadIdx.x;
  if (tid >= NT * DD) return;
  int node = tid >> 5, f = tid & 31;
  const float* hr = h + node * DD;
  float ad = 0.f, as = 0.f;
  #pragma unroll
  for (int k = 0; k < DD; ++k) {
    float hv = hr[k];
    ad += hv * Wm1[k * DD + f];
    as += hv * Wm1[(DD + k) * DD + f];
  }
  Hd[tid] = ad;
  Hs[tid] = as;
}

// tvec = b_msg1 + temb @ W_msg1[rows 65:97]   (radial is row 64)
__global__ void k_tvec(const float* __restrict__ temb, const float* __restrict__ Wm1,
                       const float* __restrict__ bm1, float* __restrict__ tvec) {
  int tid = blockIdx.x * blockDim.x + threadIdx.x;
  if (tid >= BGR * DD) return;
  int b = tid >> 5, f = tid & 31;
  float a = bm1[f];
  #pragma unroll
  for (int k = 0; k < DD; ++k) a += temb[b * DD + k] * Wm1[(65 + k) * DD + f];
  tvec[tid] = a;
}

// --------------------------- the WMMA edge kernel --------------------------
// One wave per destination node i: iterate its 128 incoming edges in 8 tiles
// of 16 (lane column = src index within tile). Per tile:
//   m1 = silu(Hd[i] + Hs[j] + radial*w_rad + tvec)          (VALU, B-layout)
//   m2 = silu(Wmsg2^T @ m1^T + b)                            (2x WMMA)
//   m  = m2 * sigmoid(w_att . m2 + b)                        (lane-local dot)
//   pw = w_pos2 . silu(Wpos1^T @ m^T + b) + b                (2x WMMA)
//   accumulate aggr_h += m ; aggr_p += pd_hat * pw
__global__ __launch_bounds__(256) void k_edge(
    const float* __restrict__ Hd, const float* __restrict__ Hs,
    const float* __restrict__ tvec, const float* __restrict__ wrad,
    const float* __restrict__ Wm2, const float* __restrict__ bm2,
    const float* __restrict__ Watt, const float* __restrict__ batt,
    const float* __restrict__ Wp1, const float* __restrict__ bp1,
    const float* __restrict__ Wp2, const float* __restrict__ bp2,
    const float* __restrict__ pin, float* __restrict__ pout,
    float* __restrict__ aggr) {
  int wave = blockIdx.x * (blockDim.x >> 5) + (threadIdx.x >> 5);
  int lane = threadIdx.x & 31;
  int lh   = lane >> 4;       // lane half
  int col  = lane & 15;       // edge column / A-matrix row
  int i    = wave;            // destination node (grid == 8192 waves exactly)
  int bg   = i >> 7;          // graph index

  int fo    = lh * 8;         // D-tile row/feature offset for this half
  int fbase = lh * 16;        // B-layout feature base (this lane's 16 features)

  // ---- per-lane layer invariants -----------------------------------------
  float hd[16], tv[16], wr[16];
  {
    const float* hdp = Hd + i * DD + fbase;
    const float* tvp = tvec + bg * DD + fbase;
    const float* wrp = wrad + fbase;
    #pragma unroll
    for (int e = 0; e < 16; ++e) { hd[e] = hdp[e]; tv[e] = tvp[e]; wr[e] = wrp[e]; }
  }

  // A operands: W^T tiles. A-layout (16-bit, 16x32), contiguous K to match B:
  //   lanes 0-15 : elements 0..15 = K0..15   (row M = lane)
  //   lanes 16-31: elements 0..15 = K16..31  (row M = lane-16)
  v16h Am2_0, Am2_1, Ap1_0, Ap1_1;
  #pragma unroll
  for (int t = 0; t < 8; ++t) {
    int k0 = fbase + t;       // K = 16h + t
    int k1 = fbase + 8 + t;   // K = 16h + 8 + t
    Am2_0[t]     = (_Float16)Wm2[k0 * DD + col];        // out rows 0..15
    Am2_0[8 + t] = (_Float16)Wm2[k1 * DD + col];
    Am2_1[t]     = (_Float16)Wm2[k0 * DD + 16 + col];   // out rows 16..31
    Am2_1[8 + t] = (_Float16)Wm2[k1 * DD + 16 + col];
    Ap1_0[t]     = (_Float16)Wp1[k0 * DD + col];
    Ap1_0[8 + t] = (_Float16)Wp1[k1 * DD + col];
    Ap1_1[t]     = (_Float16)Wp1[k0 * DD + 16 + col];
    Ap1_1[8 + t] = (_Float16)Wp1[k1 * DD + 16 + col];
  }

  // D-tile per-row constants: tile0 row r -> feature fo+r, tile1 -> 16+fo+r
  float bm2a[8], bm2b[8], wata[8], watb[8], bp1a[8], bp1b[8], wp2a[8], wp2b[8];
  #pragma unroll
  for (int r = 0; r < 8; ++r) {
    int f0 = fo + r, f1 = 16 + fo + r;
    bm2a[r] = bm2[f0];  bm2b[r] = bm2[f1];
    wata[r] = Watt[f0]; watb[r] = Watt[f1];
    bp1a[r] = bp1[f0];  bp1b[r] = bp1[f1];
    wp2a[r] = Wp2[f0];  wp2b[r] = Wp2[f1];
  }
  float battv = batt[0], bp2v = bp2[0];
  float pix = pin[i * 2], piy = pin[i * 2 + 1];

  float accM0[8] = {0.f, 0.f, 0.f, 0.f, 0.f, 0.f, 0.f, 0.f};
  float accM1[8] = {0.f, 0.f, 0.f, 0.f, 0.f, 0.f, 0.f, 0.f};
  float accPx = 0.f, accPy = 0.f;

  for (int jt = 0; jt < 8; ++jt) {
    int j = (bg << 7) + jt * 16 + col;          // this lane's source node
    if (jt < 7)                                  // gfx1250 global_prefetch_b8
      __builtin_prefetch(Hs + ((bg << 7) + (jt + 1) * 16 + col) * DD, 0, 1);

    const float* hsp = Hs + j * DD + fbase;
    float pjx = pin[j * 2], pjy = pin[j * 2 + 1];
    float pdx = pix - pjx, pdy = piy - pjy;      // p[dst] - p[src]
    float radial = pdx * pdx + pdy * pdy;
    float inv = frcp(fsqrt(radial) + 1e-6f);
    pdx *= inv; pdy *= inv;

    // B operand (M^T): lane = edge col, elements e = feature fbase+e, packed f16
    v16h bmat;
    #pragma unroll
    for (int e = 0; e < 16; ++e) {
      float m = hd[e] + hsp[e] + radial * wr[e] + tv[e];
      bmat[e] = (_Float16)siluf(m);
    }

    v8f cz = {};
    v8f c0 = wmma16(Am2_0, bmat, cz);            // features 0..15 (rows) x edges
    v8f c1 = wmma16(Am2_1, bmat, cz);            // features 16..31

    float m0[8], m1[8];
    float part = 0.f;
    #pragma unroll
    for (int r = 0; r < 8; ++r) {
      m0[r] = siluf(c0[r] + bm2a[r]);
      m1[r] = siluf(c1[r] + bm2b[r]);
      part += m0[r] * wata[r] + m1[r] * watb[r]; // partial attention dot
    }
    part += __shfl_xor(part, 16, 32);            // combine feature halves
    float att = sigmf(part + battv);

    v16h b2;                                      // gated m back to B-layout
    #pragma unroll
    for (int r = 0; r < 8; ++r) {
      m0[r] *= att; m1[r] *= att;
      accM0[r] += m0[r]; accM1[r] += m1[r];
      float o0 = __shfl_xor(m0[r], 16, 32);      // partner's feature 8+r / r
      float o1 = __shfl_xor(m1[r], 16, 32);      // partner's feature 24+r / 16+r
      // lanes<16 need features 0..15 = {own m0, partner m0}
      // lanes>=16 need features 16..31 = {partner m1, own m1}
      b2[r]     = (_Float16)(lh ? o1 : m0[r]);
      b2[8 + r] = (_Float16)(lh ? m1[r] : o0);
    }

    v8f d0 = wmma16(Ap1_0, b2, cz);
    v8f d1 = wmma16(Ap1_1, b2, cz);
    float pp = 0.f;
    #pragma unroll
    for (int r = 0; r < 8; ++r) {
      pp += siluf(d0[r] + bp1a[r]) * wp2a[r];
      pp += siluf(d1[r] + bp1b[r]) * wp2b[r];
    }
    pp += __shfl_xor(pp, 16, 32);
    float pw = pp + bp2v;
    accPx += pdx * pw;
    accPy += pdy * pw;
  }

  // ---- reduce over the 16 edge columns and write -------------------------
  #pragma unroll
  for (int r = 0; r < 8; ++r) {
    float s0 = accM0[r];
    s0 += __shfl_xor(s0, 1, 32); s0 += __shfl_xor(s0, 2, 32);
    s0 += __shfl_xor(s0, 4, 32); s0 += __shfl_xor(s0, 8, 32);
    float s1 = accM1[r];
    s1 += __shfl_xor(s1, 1, 32); s1 += __shfl_xor(s1, 2, 32);
    s1 += __shfl_xor(s1, 4, 32); s1 += __shfl_xor(s1, 8, 32);
    if (col == 0) {                               // lane 0 and lane 16 write
      aggr[i * DD + fo + r]      = s0;            // features fo..fo+7
      aggr[i * DD + 16 + fo + r] = s1;            // features 16+fo..
    }
  }
  float sx = accPx, sy = accPy;                   // both halves computed the
  sx += __shfl_xor(sx, 1, 32); sx += __shfl_xor(sx, 2, 32);  // same edges ->
  sx += __shfl_xor(sx, 4, 32); sx += __shfl_xor(sx, 8, 32);  // take half 0
  sy += __shfl_xor(sy, 1, 32); sy += __shfl_xor(sy, 2, 32);
  sy += __shfl_xor(sy, 4, 32); sy += __shfl_xor(sy, 8, 32);
  if (lane == 0) {
    pout[i * 2]     = pix + sx * (1.0f / NN);     // scatter-mean + residual
    pout[i * 2 + 1] = piy + sy * (1.0f / NN);
  }
}

// --------------------------- node / temb updates ---------------------------

__global__ void k_node(float* __restrict__ h, const float* __restrict__ aggr,
                       const float* __restrict__ Wn1, const float* __restrict__ bn1,
                       const float* __restrict__ Wn2, const float* __restrict__ bn2) {
  int node = blockIdx.x * blockDim.x + threadIdx.x;
  if (node >= NT) return;
  float hr[DD], ar[DD], hid[DD];
  #pragma unroll
  for (int k = 0; k < DD; ++k) { hr[k] = h[node * DD + k]; ar[k] = aggr[node * DD + k]; }
  #pragma unroll 4
  for (int f = 0; f < DD; ++f) {
    float a = bn1[f];
    #pragma unroll
    for (int k = 0; k < DD; ++k) a += hr[k] * Wn1[k * DD + f];
    #pragma unroll
    for (int k = 0; k < DD; ++k) a += ar[k] * Wn1[(DD + k) * DD + f];
    hid[f] = siluf(a);
  }
  #pragma unroll 4
  for (int f = 0; f < DD; ++f) {
    float u = bn2[f];
    #pragma unroll
    for (int k = 0; k < DD; ++k) u += hid[k] * Wn2[k * DD + f];
    float lr = (u > 0.f) ? u : 0.01f * u;        // LeakyReLU
    h[node * DD + f] = lr + hr[f];               // residual
  }
}

__global__ void k_temb_up(float* __restrict__ temb, const float* __restrict__ Wt,
                          const float* __restrict__ bt) {
  int b = blockIdx.x * blockDim.x + threadIdx.x;
  if (b >= BGR) return;
  float tr[DD], o[DD];
  #pragma unroll
  for (int k = 0; k < DD; ++k) tr[k] = temb[b * DD + k];
  #pragma unroll 4
  for (int f = 0; f < DD; ++f) {
    float a = bt[f];
    #pragma unroll
    for (int k = 0; k < DD; ++k) a += tr[k] * Wt[k * DD + f];
    o[f] = siluf(a);
  }
  #pragma unroll
  for (int f = 0; f < DD; ++f) temb[b * DD + f] = o[f];
}

__global__ void k_out(const float* __restrict__ p, float* __restrict__ out) {
  int tid = blockIdx.x * blockDim.x + threadIdx.x;
  if (tid >= BGR * NSH * 2) return;
  int b = tid / (NSH * 2);
  int rem = tid - b * (NSH * 2);                  // shelves are contiguous
  out[tid] = p[(b * NN + NG) * 2 + rem];
}

// ------------------------------- launcher ----------------------------------

extern "C" void kernel_launch(void* const* d_in, const int* in_sizes, int n_in,
                              void* d_out, int out_size, void* d_ws, size_t ws_size,
                              hipStream_t stream) {
  (void)in_sizes; (void)n_in; (void)out_size; (void)ws_size;
  const float* pos      = (const float*)d_in[0];
  const int*   ts       = (const int*)d_in[1];
  const float* goal_pos = (const float*)d_in[2];
  const float* W_hin    = (const float*)d_in[3];
  const float* b_hin    = (const float*)d_in[4];
  const float* W_msg1   = (const float*)d_in[5];
  const float* b_msg1   = (const float*)d_in[6];
  const float* W_msg2   = (const float*)d_in[7];
  const float* b_msg2   = (const float*)d_in[8];
  const float* W_att    = (const float*)d_in[9];
  const float* b_att    = (const float*)d_in[10];
  const float* W_pos1   = (const float*)d_in[11];
  const float* b_pos1   = (const float*)d_in[12];
  const float* W_pos2   = (const float*)d_in[13];
  const float* b_pos2   = (const float*)d_in[14];
  const float* W_node1  = (const float*)d_in[15];
  const float* b_node1  = (const float*)d_in[16];
  const float* W_node2  = (const float*)d_in[17];
  const float* b_node2  = (const float*)d_in[18];
  const float* W_t      = (const float*)d_in[19];
  const float* b_t      = (const float*)d_in[20];

  // Workspace carve-out (~4.3 MB total)
  char* ws = (char*)d_ws;
  size_t off = 0;
  auto carve = [&](size_t bytes) -> char* {
    char* p = ws + off;
    off += (bytes + 255) & ~(size_t)255;
    return p;
  };
  float* pA   = (float*)carve((size_t)NT * 2 * sizeof(float));
  float* pB   = (float*)carve((size_t)NT * 2 * sizeof(float));
  float* h    = (float*)carve((size_t)NT * DD * sizeof(float));
  float* temb = (float*)carve((size_t)BGR * DD * sizeof(float));
  float* tvec = (float*)carve((size_t)BGR * DD * sizeof(float));
  float* Hd   = (float*)carve((size_t)NT * DD * sizeof(float));
  float* Hs   = (float*)carve((size_t)NT * DD * sizeof(float));
  float* aggr = (float*)carve((size_t)NT * DD * sizeof(float));

  k_init_temb<<<(BGR * DD + 255) / 256, 256, 0, stream>>>(ts, temb);
  k_init_hp<<<(NT + 255) / 256, 256, 0, stream>>>(W_hin, b_hin, goal_pos, pos, h, pA);

  float* pin = pA;
  float* pout = pB;
  for (int l = 0; l < NL; ++l) {
    const float* Wm1l = W_msg1 + (size_t)l * 97 * DD;
    k_precomp<<<(NT * DD + 255) / 256, 256, 0, stream>>>(h, Wm1l, Hd, Hs);
    k_tvec<<<(BGR * DD + 255) / 256, 256, 0, stream>>>(temb, Wm1l, b_msg1 + l * DD, tvec);
    k_edge<<<NT / 8, 256, 0, stream>>>(Hd, Hs, tvec, Wm1l + 64 * DD,
                                       W_msg2 + (size_t)l * DD * DD, b_msg2 + l * DD,
                                       W_att + l * DD, b_att + l,
                                       W_pos1 + (size_t)l * DD * DD, b_pos1 + l * DD,
                                       W_pos2 + l * DD, b_pos2 + l,
                                       pin, pout, aggr);
    if (l < NL - 1) {
      k_node<<<(NT + 255) / 256, 256, 0, stream>>>(h, aggr,
          W_node1 + (size_t)l * 64 * DD, b_node1 + l * DD,
          W_node2 + (size_t)l * DD * DD, b_node2 + l * DD);
      k_temb_up<<<1, 64, 0, stream>>>(temb, W_t + (size_t)l * DD * DD, b_t + l * DD);
    }
    float* t = pin; pin = pout; pout = t;
  }
  k_out<<<(BGR * NSH * 2 + 255) / 256, 256, 0, stream>>>(pin, (float*)d_out);
}
